// ContrastiveTableHead_46969762349373
// MI455X (gfx1250) — matrix operand order repaired and checked by reading, not compiled, not run on hardware
//
#include <hip/hip_runtime.h>

#define BB   4096
#define DD   1024
#define RR   30
#define NOUT 63
#define NCOL 3072   // packed output columns: book | change | cancel

typedef __attribute__((ext_vector_type(16))) __bf16 v16bf;
typedef __attribute__((ext_vector_type(8)))  __bf16 v8bf;
typedef __attribute__((ext_vector_type(8)))  float  v8f;

__device__ __forceinline__ unsigned short f32_to_bf16_rne(float f) {
  unsigned int u = __float_as_uint(f);
  u += 0x7fffu + ((u >> 16) & 1u);
  return (unsigned short)(u >> 16);
}

__global__ __launch_bounds__(256) void cvt_f32_bf16(const float* __restrict__ src,
                                                    unsigned short* __restrict__ dst,
                                                    int n) {
  int i = blockIdx.x * 256 + threadIdx.x;
  if (i < n) dst[i] = f32_to_bf16_rne(src[i]);
}

__global__ __launch_bounds__(256) void pack_bias(const float* __restrict__ b0,
                                                 const float* __restrict__ b1,
                                                 const float* __restrict__ b2,
                                                 float* __restrict__ out) {
  int i = blockIdx.x * 256 + threadIdx.x;
  if (i >= NCOL) return;
  out[i] = (i < DD) ? b0[i] : (i < 2 * DD) ? b1[i - DD] : b2[i - 2 * DD];
}

// ---- WMMA GEMM helpers: double-buffered fragment pipeline -------------------

struct Frag {
  v8bf a0, a1;       // A: lane's K runs [k+8*hi, +7] and [k+8*hi+16, +23]
  v8bf b[4][2];      // B: per N-tile, K runs [k+8*hi, +7] and [k+8*hi+16, +23]
};

__device__ __forceinline__ void load_frag(Frag& f,
                                          const unsigned short* __restrict__ arow,
                                          const unsigned short* __restrict__ brow0,
                                          int k) {
  f.a0 = *(const v8bf*)(arow + k);
  f.a1 = *(const v8bf*)(arow + k + 16);
  #pragma unroll
  for (int t = 0; t < 4; ++t) {
    const unsigned short* bp = brow0 + (size_t)t * 16 * DD + k;
    f.b[t][0] = *(const v8bf*)(bp);
    f.b[t][1] = *(const v8bf*)(bp + 16);
  }
}

__device__ __forceinline__ void mma_frag(const Frag& f, v8f acc[4]) {
  v16bf av = __builtin_shufflevector(f.a0, f.a1, 0,1,2,3,4,5,6,7,8,9,10,11,12,13,14,15);
  #pragma unroll
  for (int t = 0; t < 4; ++t) {
    v16bf bv = __builtin_shufflevector(f.b[t][0], f.b[t][1],
                                       0,1,2,3,4,5,6,7,8,9,10,11,12,13,14,15);
    acc[t] = __builtin_amdgcn_wmma_f32_16x16x32_bf16(false, av, false, bv,
                                                     (short)0, acc[t], false, false);
  }
}

// E[4096x3072] = Cbf16[4096x1024] @ Wcat[3072x1024]^T + biascat
// One wave computes a 16x64 output slab (4 N-tiles sharing the A fragment).
__global__ __launch_bounds__(32) void gemm_wmma_bf16(
    const unsigned short* __restrict__ A,   // (4096 x 1024) bf16 row-major
    const unsigned short* __restrict__ Bw,  // (3072 x 1024) bf16; row n = output col n
    const float* __restrict__ bias,         // (3072)
    float* __restrict__ E) {                // (4096 x 3072) f32
  const int lane = threadIdx.x & 31;
  const int m    = lane & 15;
  const int hi   = lane >> 4;               // 0 or 1
  const int tm   = blockIdx.y;              // 256 M tiles
  const int tn4  = blockIdx.x;              // 48 groups of 4 N tiles

  const unsigned short* arow  = A + (size_t)(tm * 16 + m) * DD + hi * 8;
  const unsigned short* brow0 = Bw + (size_t)(tn4 * 64 + m) * DD + hi * 8;

  v8f acc[4] = {{}, {}, {}, {}};
  Frag fA, fB;

  load_frag(fA, arow, brow0, 0);
  for (int k = 0; k < DD; k += 64) {
    load_frag(fB, arow, brow0, k + 32);          // prefetch odd k-step
    mma_frag(fA, acc);                           // consume even k-step
    load_frag(fA, arow, brow0, (k + 64) & (DD - 1));  // prefetch next even (wraps on last)
    mma_frag(fB, acc);
  }

  // D layout: lane l, vgpr v -> row v + 8*hi, col l&15
  const int colbase = tn4 * 64 + m;
  const int rowbase = tm * 16 + hi * 8;
  #pragma unroll
  for (int t = 0; t < 4; ++t) {
    const int col = colbase + t * 16;
    const float bv = bias[col];
    #pragma unroll
    for (int v = 0; v < 8; ++v) {
      E[(size_t)(rowbase + v) * NCOL + col] = acc[t][v] + bv;
    }
  }
}

// One block per batch row: flight logits, goal head, gates, softmaxes, assembly.
__global__ __launch_bounds__(256) void fused_head(
    const float* __restrict__ constraint,   // (B,D)
    const float* __restrict__ table,        // (B,R,D)
    const float* __restrict__ no_flight,    // (B,D)
    const float* __restrict__ res,          // (B,D)
    const float* __restrict__ goalW,        // (3,D)
    const float* __restrict__ goalb,        // (3)
    const float* __restrict__ cgW,          // change gate (2D)
    const float* __restrict__ cgb,          // (1)
    const float* __restrict__ kgW,          // cancel gate (2D)
    const float* __restrict__ kgb,          // (1)
    const float* __restrict__ E,            // (B,3072)
    float* __restrict__ out) {              // (B,63)
  const int b    = blockIdx.x;
  const int tid  = threadIdx.x;
  const int lane = tid & 31;
  const int wave = tid >> 5;                // 8 waves

  __shared__ float qb[DD];
  __shared__ float qc[DD];
  __shared__ float red[72];
  __shared__ float pb[RR + 1];
  __shared__ float pc[RR + 1];
  __shared__ float fin[8];

  const float* Eb = E + (size_t)b * NCOL;
  for (int i = tid; i < DD; i += 256) {
    qb[i] = Eb[i];            // book_emb
    qc[i] = Eb[DD + i];       // change_emb
  }
  __syncthreads();

  // tasks: 0..30 book logits, 31..61 change logits, 62..64 goal, 65 change gate, 66 cancel gate
  for (int task = wave; task < 67; task += 8) {
    float s = 0.0f;
    if (task < 62) {
      const int ch = (task < 31) ? 0 : 1;
      const int r  = ch ? (task - 31) : task;
      const float* row = (r == 0) ? (no_flight + (size_t)b * DD)
                                  : (table + ((size_t)b * RR + (r - 1)) * DD);
      const float* q = ch ? qc : qb;
      for (int i = lane * 4; i < DD; i += 128) {
        float4 t = *(const float4*)(row + i);
        s += t.x * q[i] + t.y * q[i + 1] + t.z * q[i + 2] + t.w * q[i + 3];
      }
    } else if (task < 65) {
      const float* w = goalW + (size_t)(task - 62) * DD;
      const float* c = constraint + (size_t)b * DD;
      for (int i = lane * 4; i < DD; i += 128) {
        float4 t = *(const float4*)(w + i);
        float4 u = *(const float4*)(c + i);
        s += t.x * u.x + t.y * u.y + t.z * u.z + t.w * u.w;
      }
    } else {
      const float* w  = (task == 65) ? cgW : kgW;
      const float* e2 = res + (size_t)b * DD;
      if (task == 65) {
        for (int i = lane * 4; i < DD; i += 128) {
          float4 t = *(const float4*)(w + i);
          s += t.x * qc[i] + t.y * qc[i + 1] + t.z * qc[i + 2] + t.w * qc[i + 3];
        }
      } else {
        const float* e1 = Eb + 2 * DD;   // cancel_emb
        for (int i = lane * 4; i < DD; i += 128) {
          float4 t = *(const float4*)(w + i);
          float4 u = *(const float4*)(e1 + i);
          s += t.x * u.x + t.y * u.y + t.z * u.z + t.w * u.w;
        }
      }
      for (int i = lane * 4; i < DD; i += 128) {
        float4 t = *(const float4*)(w + DD + i);
        float4 u = *(const float4*)(e2 + i);
        s += t.x * u.x + t.y * u.y + t.z * u.z + t.w * u.w;
      }
    }
    #pragma unroll
    for (int off = 16; off > 0; off >>= 1) s += __shfl_xor(s, off, 32);
    if (lane == 0) red[task] = s;
  }
  __syncthreads();

  if (tid == 0) {                      // book softmax over 31 (scaled 1/sqrt(D))
    float mx = -1e30f;
    for (int r = 0; r <= RR; ++r) { float l = red[r] * 0.03125f; pb[r] = l; mx = fmaxf(mx, l); }
    float sum = 0.0f;
    for (int r = 0; r <= RR; ++r) { float e = __expf(pb[r] - mx); pb[r] = e; sum += e; }
    float inv = 1.0f / sum;
    for (int r = 0; r <= RR; ++r) pb[r] *= inv;
  } else if (tid == 32) {              // change softmax over 31
    float mx = -1e30f;
    for (int r = 0; r <= RR; ++r) { float l = red[31 + r] * 0.03125f; pc[r] = l; mx = fmaxf(mx, l); }
    float sum = 0.0f;
    for (int r = 0; r <= RR; ++r) { float e = __expf(pc[r] - mx); pc[r] = e; sum += e; }
    float inv = 1.0f / sum;
    for (int r = 0; r <= RR; ++r) pc[r] *= inv;
  } else if (tid == 64) {              // goal softmax + gates
    float l0 = red[62] + goalb[0], l1 = red[63] + goalb[1], l2 = red[64] + goalb[2];
    float mx = fmaxf(l0, fmaxf(l1, l2));
    float e0 = __expf(l0 - mx), e1 = __expf(l1 - mx), e2 = __expf(l2 - mx);
    float inv = 1.0f / (e0 + e1 + e2);
    fin[0] = e0 * inv;  // book_goal
    fin[1] = e1 * inv;  // change_goal
    fin[2] = e2 * inv;  // cancel_goal
    fin[3] = 1.0f / (1.0f + __expf(-(red[65] + cgb[0])));  // change_prob
    fin[4] = 1.0f / (1.0f + __expf(-(red[66] + kgb[0])));  // cancel_prob
  }
  __syncthreads();

  if (tid < NOUT) {
    const float bg = fin[0], cg = fin[1], xg = fin[2];
    const float cp = fin[3], xp = fin[4];
    float v;
    if (tid == 0)      v = bg * pb[0] + cg * cp * pc[0];
    else if (tid == 1) v = xg * xp;
    else if (tid == 2) v = xg * (1.0f - xp) + cg * (1.0f - cp);
    else if (tid < 33) v = bg * pb[tid - 2];           // tid 3..32 -> pb[1..30]
    else               v = cg * cp * pc[tid - 32];     // tid 33..62 -> pc[1..30]
    out[(size_t)b * NOUT + tid] = v;
  }
}

extern "C" void kernel_launch(void* const* d_in, const int* in_sizes, int n_in,
                              void* d_out, int out_size, void* d_ws, size_t ws_size,
                              hipStream_t stream) {
  const float* constraint = (const float*)d_in[0];
  const float* table      = (const float*)d_in[1];
  const float* no_flight  = (const float*)d_in[2];
  const float* res        = (const float*)d_in[3];
  const float* goalW      = (const float*)d_in[4];
  const float* goalb      = (const float*)d_in[5];
  const float* bookW      = (const float*)d_in[6];
  const float* bookb      = (const float*)d_in[7];
  const float* changeW    = (const float*)d_in[8];
  const float* changeb    = (const float*)d_in[9];
  const float* cancelW    = (const float*)d_in[10];
  const float* cancelb    = (const float*)d_in[11];
  const float* cgW        = (const float*)d_in[12];
  const float* cgb        = (const float*)d_in[13];
  const float* kgW        = (const float*)d_in[14];
  const float* kgb        = (const float*)d_in[15];

  char* ws = (char*)d_ws;
  unsigned short* Cb = (unsigned short*)(ws);                           // 8 MB  bf16 activations
  unsigned short* Wc = (unsigned short*)(ws + ((size_t)8  << 20));      // 6 MB  bf16 packed weights
  float* biascat     = (float*)(ws + ((size_t)14 << 20));               // 12 KB packed bias
  float* E           = (float*)(ws + ((size_t)14 << 20) + (64u << 10)); // 48 MB embeddings

  const int nC = BB * DD;
  const int nW = DD * DD;
  cvt_f32_bf16<<<(nC + 255) / 256, 256, 0, stream>>>(constraint, Cb, nC);
  cvt_f32_bf16<<<(nW + 255) / 256, 256, 0, stream>>>(bookW,   Wc,          nW);
  cvt_f32_bf16<<<(nW + 255) / 256, 256, 0, stream>>>(changeW, Wc + nW,     nW);
  cvt_f32_bf16<<<(nW + 255) / 256, 256, 0, stream>>>(cancelW, Wc + 2 * nW, nW);
  pack_bias<<<(NCOL + 255) / 256, 256, 0, stream>>>(bookb, changeb, cancelb, biascat);

  gemm_wmma_bf16<<<dim3(48, BB / 16), 32, 0, stream>>>(Cb, Wc, biascat, E);

  fused_head<<<BB, 256, 0, stream>>>(constraint, table, no_flight, res,
                                     goalW, goalb, cgW, cgb, kgW, kgb,
                                     E, (float*)d_out);
}
